// NMR2MolDecoder_25486335934573
// MI455X (gfx1250) — matrix operand order repaired and checked by reading, not compiled, not run on hardware
//
#include <hip/hip_runtime.h>
#include <math.h>

// ---------------- model constants ----------------
#define D_MODEL 512
#define D_FF    2048
#define NHEAD   8
#define NLAYERS 4
#define VOCABSZ 512
#define BSZ     16
#define LFULL   513
#define LT      512   // target length after dropping last token
#define SRCLEN  256

typedef __attribute__((ext_vector_type(16))) __bf16       v16bf;
typedef __attribute__((ext_vector_type(8)))  float        v8f;
typedef __attribute__((ext_vector_type(8)))  unsigned int v8u;

__device__ __forceinline__ unsigned short bfbits(float f) {
  __bf16 h = (__bf16)f;                       // native bf16 convert on gfx1250
  return __builtin_bit_cast(unsigned short, h);
}
__device__ __forceinline__ unsigned int packbf(float lo, float hi) {
  return (unsigned int)bfbits(lo) | ((unsigned int)bfbits(hi) << 16);
}

// ---- WMMA fragment loaders (ISA §7.12.2 layouts) ----
__device__ __forceinline__ v16bf fragA(const unsigned short* lsA, int row0, int lidx, int hlf) {
  const unsigned short* pa = lsA + (row0 + lidx) * 32;
  v8u u;
  #pragma unroll
  for (int j = 0; j < 4; ++j) {
    u[j]     = *(const unsigned int*)(pa + hlf * 8 + 2 * j);        // K 0..15 half
    u[j + 4] = *(const unsigned int*)(pa + 16 + hlf * 8 + 2 * j);   // K 16..31 half
  }
  return __builtin_bit_cast(v16bf, u);
}
__device__ __forceinline__ v16bf fragB(const unsigned short* lsB, int n0, int lidx, int hlf) {
  const unsigned short* pb = lsB + (n0 + lidx) * 32;
  v8u u;
  #pragma unroll
  for (int j = 0; j < 4; ++j) {
    u[j]     = *(const unsigned int*)(pb + hlf * 8 + 2 * j);
    u[j + 4] = *(const unsigned int*)(pb + 16 + hlf * 8 + 2 * j);
  }
  return __builtin_bit_cast(v16bf, u);
}

// =====================================================================
// Generic batched GEMM:  C = alpha * A @ B(^T) + bias (+ residual) (relu?)
// A: M x K (row major, stride lda)
// B: transB==0 -> K x N (stride ldb);  transB==1 -> N x K (stride ldb)
// Batched over gridDim.z with split offsets: z -> (z/Hd, z%Hd)
// Block: 256 threads = 8 waves; tile 128x64; each wave owns 32x32
// (2 A-frags x 2 B-frags -> 4 WMMAs/K-step, fragment reuse x2).
// Double-buffered LDS with register prefetch so global_load latency
// overlaps the v_wmma_f32_16x16x32_bf16 issue.
// Requirements: M%128==0, N%64==0, K%32==0 (true for all shapes here).
// =====================================================================
__global__ __launch_bounds__(256)
void gemm_wmma_bf16(const float* __restrict__ A, const float* __restrict__ B,
                    const float* __restrict__ bias, const float* __restrict__ residual,
                    float* __restrict__ C,
                    int M, int N, int K, int lda, int ldb, int ldc,
                    int Hd,
                    long long sAb, long long sAh, long long sBb, long long sBh,
                    long long sCb, long long sCh,
                    float alpha, int transB, int relu)
{
  __shared__ unsigned short lsA[2][128 * 32];   // [m][k] bf16, double buffered
  __shared__ unsigned short lsB[2][64 * 32];    // [n][k] bf16, double buffered

  const int z  = blockIdx.z;
  const int zb = z / Hd, zh = z % Hd;
  A += (long long)zb * sAb + (long long)zh * sAh;
  B += (long long)zb * sBb + (long long)zh * sBh;
  const long long offC = (long long)zb * sCb + (long long)zh * sCh;
  float* Cp = C + offC;
  const float* Rp = residual ? residual + offC : (const float*)0;

  const int t    = threadIdx.x;
  const int lane = t & 31;
  const int w    = t >> 5;      // 0..7
  const int wm   = w & 3;       // wave row group (32 rows each)
  const int wn   = w >> 2;      // wave col group (32 cols each)
  const int blockM = blockIdx.y * 128;
  const int blockN = blockIdx.x * 64;

  const int hlf  = lane >> 4;   // lane half (0/1)
  const int lidx = lane & 15;

  // staging assignments
  const int rowA = t >> 1;          // 0..127
  const int colA = (t & 1) * 16;    // 0 / 16
  const int rowB = t >> 2;          // 0..63   (transB path)
  const int colB = (t & 3) * 8;     // 0..24
  const int krB  = t >> 3;          // 0..31   (normal path)
  const int ncB  = (t & 7) * 8;     // 0..56

  v8f acc[2][2] = {};
  float4 ra[4];    // 16 floats of A tile
  float4 rb[2];    // 8 floats of B tile

  // ---- prefetch helpers (to registers) ----
  auto loadRegs = [&](int kk) {
    const float* ga = A + (long long)(blockM + rowA) * lda + kk + colA;
    #pragma unroll
    for (int j = 0; j < 4; ++j) ra[j] = *(const float4*)(ga + 4 * j);
    if (transB) {
      const float* gb = B + (long long)(blockN + rowB) * ldb + kk + colB;
      rb[0] = *(const float4*)(gb);
      rb[1] = *(const float4*)(gb + 4);
    } else {
      const float* gb = B + (long long)(kk + krB) * ldb + blockN + ncB;
      rb[0] = *(const float4*)(gb);
      rb[1] = *(const float4*)(gb + 4);
    }
  };
  // ---- convert + stage registers into LDS buffer `buf` ----
  auto stageRegs = [&](int buf) {
    unsigned int o[8];
    #pragma unroll
    for (int j = 0; j < 4; ++j) {
      o[2 * j]     = packbf(ra[j].x, ra[j].y);
      o[2 * j + 1] = packbf(ra[j].z, ra[j].w);
    }
    unsigned int* la = (unsigned int*)&lsA[buf][rowA * 32 + colA];
    *(uint4*)(la)     = make_uint4(o[0], o[1], o[2], o[3]);
    *(uint4*)(la + 4) = make_uint4(o[4], o[5], o[6], o[7]);
    if (transB) {
      unsigned int b0 = packbf(rb[0].x, rb[0].y);
      unsigned int b1 = packbf(rb[0].z, rb[0].w);
      unsigned int b2 = packbf(rb[1].x, rb[1].y);
      unsigned int b3 = packbf(rb[1].z, rb[1].w);
      *(uint4*)&lsB[buf][rowB * 32 + colB] = make_uint4(b0, b1, b2, b3);
    } else {
      float fv[8] = { rb[0].x, rb[0].y, rb[0].z, rb[0].w,
                      rb[1].x, rb[1].y, rb[1].z, rb[1].w };
      #pragma unroll
      for (int i = 0; i < 8; ++i)
        lsB[buf][(ncB + i) * 32 + krB] = bfbits(fv[i]);
    }
  };
  // ---- 4 WMMAs from buffer `buf` ----
  auto compute = [&](int buf) {
    v16bf a0 = fragA(lsA[buf], wm * 32,      lidx, hlf);
    v16bf a1 = fragA(lsA[buf], wm * 32 + 16, lidx, hlf);
    v16bf b0 = fragB(lsB[buf], wn * 32,      lidx, hlf);
    v16bf b1 = fragB(lsB[buf], wn * 32 + 16, lidx, hlf);
    acc[0][0] = __builtin_amdgcn_wmma_f32_16x16x32_bf16(false, a0, false, b0, (short)0, acc[0][0], false, false);
    acc[0][1] = __builtin_amdgcn_wmma_f32_16x16x32_bf16(false, a0, false, b1, (short)0, acc[0][1], false, false);
    acc[1][0] = __builtin_amdgcn_wmma_f32_16x16x32_bf16(false, a1, false, b0, (short)0, acc[1][0], false, false);
    acc[1][1] = __builtin_amdgcn_wmma_f32_16x16x32_bf16(false, a1, false, b1, (short)0, acc[1][1], false, false);
  };

  // ---- pipelined main loop ----
  const int nk = K >> 5;
  loadRegs(0);
  stageRegs(0);
  int cur = 0;
  for (int s = 1; s < nk; ++s) {
    __syncthreads();              // buf[cur] staged & previous reads drained
    loadRegs(s * 32);             // issue next tile's global loads
    compute(cur);                 // matrix ops overlap the loads
    stageRegs(cur ^ 1);           // loadcnt wait lands here, after WMMAs
    cur ^= 1;
  }
  __syncthreads();
  compute(cur);

  // ---- epilogue: C layout lane=N, vgpr r -> M = r + 8*half ----
  #pragma unroll
  for (int fa = 0; fa < 2; ++fa) {
    #pragma unroll
    for (int fb = 0; fb < 2; ++fb) {
      const int gn = blockN + wn * 32 + fb * 16 + lidx;
      const float bv = bias ? bias[gn] : 0.0f;
      #pragma unroll
      for (int r = 0; r < 8; ++r) {
        const int gm = blockM + wm * 32 + fa * 16 + hlf * 8 + r;
        const long long off = (long long)gm * ldc + gn;
        float val = acc[fa][fb][r] * alpha + bv;
        if (Rp) val += Rp[off];
        if (relu) val = fmaxf(val, 0.0f);
        Cp[off] = val;
      }
    }
  }
}

// =====================================================================
// Embedding * sqrt(D) + sinusoidal positional encoding
// =====================================================================
__global__ void embed_pe(const int* __restrict__ ids, const float* __restrict__ emb,
                         float* __restrict__ x)
{
  long long i = (long long)blockIdx.x * blockDim.x + threadIdx.x;
  if (i >= (long long)BSZ * LT * D_MODEL) return;
  int d = (int)(i & (D_MODEL - 1));
  long long tok = i >> 9;
  int tpos = (int)(tok % LT);
  int b    = (int)(tok / LT);
  int id = ids[b * LFULL + tpos];
  int ih = d >> 1;
  float freq = __expf(-(float)(2 * ih) * 0.0179889462f);  // ln(10000)/512
  float ang  = (float)tpos * freq;
  float pe   = (d & 1) ? __cosf(ang) : __sinf(ang);
  x[i] = emb[(long long)id * D_MODEL + d] * 22.6274169979695f + pe;
}

// =====================================================================
// LayerNorm over D=512, ddof=1 variance, eps added to std (matches ref)
// =====================================================================
__global__ __launch_bounds__(128)
void layernorm512(const float* __restrict__ x, const float* __restrict__ ga,
                  const float* __restrict__ gb, float* __restrict__ out)
{
  const long long row = blockIdx.x;
  const float* xr = x + row * D_MODEL;
  float v[4]; float s = 0.f, s2 = 0.f;
  #pragma unroll
  for (int i = 0; i < 4; ++i) {
    v[i] = xr[threadIdx.x + i * 128];
    s += v[i]; s2 += v[i] * v[i];
  }
  __shared__ float rs[128], rs2[128];
  rs[threadIdx.x] = s; rs2[threadIdx.x] = s2; __syncthreads();
  for (int st = 64; st > 0; st >>= 1) {
    if (threadIdx.x < st) { rs[threadIdx.x] += rs[threadIdx.x + st]; rs2[threadIdx.x] += rs2[threadIdx.x + st]; }
    __syncthreads();
  }
  const float mean = rs[0] * (1.0f / 512.0f);
  const float var  = (rs2[0] - 512.0f * mean * mean) * (1.0f / 511.0f);
  const float inv  = 1.0f / (sqrtf(fmaxf(var, 0.0f)) + 1e-6f);
  #pragma unroll
  for (int i = 0; i < 4; ++i) {
    int d = threadIdx.x + i * 128;
    out[row * D_MODEL + d] = ga[d] * (v[i] - mean) * inv + gb[d];
  }
}

// =====================================================================
// In-place mask (-10000 fill) + softmax over rowlen (128-thread block/row)
// causal!=0: valid = (c <= qpos) && mask[b*mstride + c]  (self-attn)
// causal==0: valid = mask[b*mstride + c]                  (cross-attn)
// =====================================================================
__global__ __launch_bounds__(128)
void masksoftmax(float* __restrict__ s, const int* __restrict__ mask,
                 int rowlen, int Lt, int H, int mstride, int causal)
{
  const long long row = blockIdx.x;
  const int qpos = (int)(row % Lt);
  const int b    = (int)((row / Lt) / H);
  float* sr = s + row * (long long)rowlen;
  const int nper = rowlen >> 7;          // 4 (len 512) or 2 (len 256)
  float mv[4];
  float mx = -3.0e38f;
  for (int i = 0; i < nper; ++i) {
    int c = threadIdx.x + i * 128;
    float vs = sr[c];
    bool valid = mask[b * mstride + c] != 0;
    if (causal) valid = valid && (c <= qpos);
    vs = valid ? vs : -10000.0f;
    mv[i] = vs;
    mx = fmaxf(mx, vs);
  }
  __shared__ float red[128];
  red[threadIdx.x] = mx; __syncthreads();
  for (int st = 64; st > 0; st >>= 1) {
    if (threadIdx.x < st) red[threadIdx.x] = fmaxf(red[threadIdx.x], red[threadIdx.x + st]);
    __syncthreads();
  }
  mx = red[0]; __syncthreads();
  float sm = 0.f;
  for (int i = 0; i < nper; ++i) { mv[i] = __expf(mv[i] - mx); sm += mv[i]; }
  red[threadIdx.x] = sm; __syncthreads();
  for (int st = 64; st > 0; st >>= 1) {
    if (threadIdx.x < st) red[threadIdx.x] += red[threadIdx.x + st];
    __syncthreads();
  }
  const float inv = 1.0f / red[0];
  for (int i = 0; i < nper; ++i) sr[threadIdx.x + i * 128] = mv[i] * inv;
}

// =====================================================================
// In-place log_softmax over 512 vocab (128-thread block per row)
// =====================================================================
__global__ __launch_bounds__(128)
void logsoftmax512(float* __restrict__ s)
{
  const long long row = blockIdx.x;
  float* sr = s + row * (long long)VOCABSZ;
  float v[4]; float mx = -3.0e38f;
  #pragma unroll
  for (int i = 0; i < 4; ++i) { v[i] = sr[threadIdx.x + i * 128]; mx = fmaxf(mx, v[i]); }
  __shared__ float red[128];
  red[threadIdx.x] = mx; __syncthreads();
  for (int st = 64; st > 0; st >>= 1) {
    if (threadIdx.x < st) red[threadIdx.x] = fmaxf(red[threadIdx.x], red[threadIdx.x + st]);
    __syncthreads();
  }
  mx = red[0]; __syncthreads();
  float sm = 0.f;
  #pragma unroll
  for (int i = 0; i < 4; ++i) sm += __expf(v[i] - mx);
  red[threadIdx.x] = sm; __syncthreads();
  for (int st = 64; st > 0; st >>= 1) {
    if (threadIdx.x < st) red[threadIdx.x] += red[threadIdx.x + st];
    __syncthreads();
  }
  const float lse = logf(red[0]);
  #pragma unroll
  for (int i = 0; i < 4; ++i) sr[threadIdx.x + i * 128] = v[i] - mx - lse;
}

// =====================================================================
// host-side launch helper
// =====================================================================
static inline void launch_gemm(hipStream_t st, const float* A, const float* B,
                               const float* bias, const float* res, float* C,
                               int M, int N, int K, int lda, int ldb, int ldc,
                               int batch, int Hd,
                               long long sAb, long long sAh, long long sBb, long long sBh,
                               long long sCb, long long sCh,
                               float alpha, int transB, int relu)
{
  dim3 g((unsigned)(N / 64), (unsigned)(M / 128), (unsigned)batch);
  gemm_wmma_bf16<<<g, dim3(256), 0, st>>>(A, B, bias, res, C, M, N, K, lda, ldb, ldc,
                                          Hd, sAb, sAh, sBb, sBh, sCb, sCh,
                                          alpha, transB, relu);
}

extern "C" void kernel_launch(void* const* d_in, const int* in_sizes, int n_in,
                              void* d_out, int out_size, void* d_ws, size_t ws_size,
                              hipStream_t stream)
{
  (void)in_sizes; (void)n_in; (void)out_size; (void)ws_size;
  const int*   ids      = (const int*)  d_in[0];
  const int*   att_mask = (const int*)  d_in[1];
  const float* src      = (const float*)d_in[2];
  const int*   src_mask = (const int*)  d_in[3];
  const float* emb      = (const float*)d_in[4];
  const float* sa_W     = (const float*)d_in[5];
  const float* sa_b     = (const float*)d_in[6];
  const float* ca_W     = (const float*)d_in[7];
  const float* ca_b     = (const float*)d_in[8];
  const float* ln_a     = (const float*)d_in[9];
  const float* ln_b     = (const float*)d_in[10];
  const float* fw1      = (const float*)d_in[11];
  const float* fb1      = (const float*)d_in[12];
  const float* fw2      = (const float*)d_in[13];
  const float* fb2      = (const float*)d_in[14];
  const float* fina     = (const float*)d_in[15];
  const float* finb     = (const float*)d_in[16];
  const float* lmW      = (const float*)d_in[17];
  const float* lmb      = (const float*)d_in[18];
  float* out = (float*)d_out;

  const long long NT = (long long)BSZ * LT;      // 8192 target tokens
  const long long DD = (long long)D_MODEL * D_MODEL;

  // workspace carve-up (x,h,q,k,v 16MB each + 128MB scores/ffn-mid = 208MB)
  char* ws = (char*)d_ws;
  float* x  = (float*)ws;  ws += NT * D_MODEL * sizeof(float);
  float* h  = (float*)ws;  ws += NT * D_MODEL * sizeof(float);
  float* q  = (float*)ws;  ws += NT * D_MODEL * sizeof(float);
  float* kb = (float*)ws;  ws += NT * D_MODEL * sizeof(float);
  float* vb = (float*)ws;  ws += NT * D_MODEL * sizeof(float);
  float* sb = (float*)ws;  // B*H*Lt*Lt floats (128MB), reused for FFN mid
  float* ctx = q;          // q dead after scores -> reuse as attention output

  // ---- embedding + positional encoding ----
  {
    long long n = NT * D_MODEL;
    embed_pe<<<dim3((unsigned)((n + 255) / 256)), dim3(256), 0, stream>>>(ids, emb, x);
  }

  for (int i = 0; i < NLAYERS; ++i) {
    const float* saWi = sa_W + (long long)i * 4 * DD;
    const float* sabi = sa_b + (long long)i * 4 * D_MODEL;
    const float* caWi = ca_W + (long long)i * 4 * DD;
    const float* cabi = ca_b + (long long)i * 4 * D_MODEL;

    // ================= self attention =================
    layernorm512<<<dim3((unsigned)NT), dim3(128), 0, stream>>>(
        x, ln_a + (i * 3 + 0) * D_MODEL, ln_b + (i * 3 + 0) * D_MODEL, h);
    launch_gemm(stream, h, saWi + 0 * DD, sabi + 0 * D_MODEL, 0, q,
                8192, 512, 512, 512, 512, 512, 1, 1, 0,0,0,0,0,0, 1.f, 0, 0);
    launch_gemm(stream, h, saWi + 1 * DD, sabi + 1 * D_MODEL, 0, kb,
                8192, 512, 512, 512, 512, 512, 1, 1, 0,0,0,0,0,0, 1.f, 0, 0);
    launch_gemm(stream, h, saWi + 2 * DD, sabi + 2 * D_MODEL, 0, vb,
                8192, 512, 512, 512, 512, 512, 1, 1, 0,0,0,0,0,0, 1.f, 0, 0);
    // scores = Q @ K^T / 8   per (b,h):  512x512, K=64
    launch_gemm(stream, q, kb, 0, 0, sb,
                512, 512, 64, 512, 512, 512,
                BSZ * NHEAD, NHEAD,
                (long long)LT * D_MODEL, 64,
                (long long)LT * D_MODEL, 64,
                (long long)NHEAD * LT * LT, (long long)LT * LT,
                0.125f, 1, 0);
    masksoftmax<<<dim3((unsigned)(BSZ * NHEAD * LT)), dim3(128), 0, stream>>>(
        sb, att_mask, LT, LT, NHEAD, LFULL, 1);
    // ctx = P @ V  per (b,h): 512x64, K=512
    launch_gemm(stream, sb, vb, 0, 0, ctx,
                512, 64, 512, 512, 512, 512,
                BSZ * NHEAD, NHEAD,
                (long long)NHEAD * LT * LT, (long long)LT * LT,
                (long long)LT * D_MODEL, 64,
                (long long)LT * D_MODEL, 64,
                1.f, 0, 0);
    // x += ctx @ Wo + bo
    launch_gemm(stream, ctx, saWi + 3 * DD, sabi + 3 * D_MODEL, x, x,
                8192, 512, 512, 512, 512, 512, 1, 1, 0,0,0,0,0,0, 1.f, 0, 0);

    // ================= cross attention =================
    layernorm512<<<dim3((unsigned)NT), dim3(128), 0, stream>>>(
        x, ln_a + (i * 3 + 1) * D_MODEL, ln_b + (i * 3 + 1) * D_MODEL, h);
    launch_gemm(stream, h, caWi + 0 * DD, cabi + 0 * D_MODEL, 0, q,
                8192, 512, 512, 512, 512, 512, 1, 1, 0,0,0,0,0,0, 1.f, 0, 0);
    launch_gemm(stream, src, caWi + 1 * DD, cabi + 1 * D_MODEL, 0, kb,
                4096, 512, 512, 512, 512, 512, 1, 1, 0,0,0,0,0,0, 1.f, 0, 0);
    launch_gemm(stream, src, caWi + 2 * DD, cabi + 2 * D_MODEL, 0, vb,
                4096, 512, 512, 512, 512, 512, 1, 1, 0,0,0,0,0,0, 1.f, 0, 0);
    launch_gemm(stream, q, kb, 0, 0, sb,
                512, 256, 64, 512, 512, 256,
                BSZ * NHEAD, NHEAD,
                (long long)LT * D_MODEL, 64,
                (long long)SRCLEN * D_MODEL, 64,
                (long long)NHEAD * LT * SRCLEN, (long long)LT * SRCLEN,
                0.125f, 1, 0);
    masksoftmax<<<dim3((unsigned)(BSZ * NHEAD * LT)), dim3(128), 0, stream>>>(
        sb, src_mask, SRCLEN, LT, NHEAD, SRCLEN, 0);
    launch_gemm(stream, sb, vb, 0, 0, ctx,
                512, 64, 256, 256, 512, 512,
                BSZ * NHEAD, NHEAD,
                (long long)NHEAD * LT * SRCLEN, (long long)LT * SRCLEN,
                (long long)SRCLEN * D_MODEL, 64,
                (long long)LT * D_MODEL, 64,
                1.f, 0, 0);
    launch_gemm(stream, ctx, caWi + 3 * DD, cabi + 3 * D_MODEL, x, x,
                8192, 512, 512, 512, 512, 512, 1, 1, 0,0,0,0,0,0, 1.f, 0, 0);

    // ================= FFN =================
    layernorm512<<<dim3((unsigned)NT), dim3(128), 0, stream>>>(
        x, ln_a + (i * 3 + 2) * D_MODEL, ln_b + (i * 3 + 2) * D_MODEL, h);
    launch_gemm(stream, h, fw1 + (long long)i * D_MODEL * D_FF, fb1 + (long long)i * D_FF, 0, sb,
                8192, 2048, 512, 512, 2048, 2048, 1, 1, 0,0,0,0,0,0, 1.f, 0, 1);
    launch_gemm(stream, sb, fw2 + (long long)i * D_FF * D_MODEL, fb2 + (long long)i * D_MODEL, x, x,
                8192, 512, 2048, 2048, 512, 512, 1, 1, 0,0,0,0,0,0, 1.f, 0, 0);
  }

  // ---- final LN + LM head + log_softmax ----
  layernorm512<<<dim3((unsigned)NT), dim3(128), 0, stream>>>(x, fina, finb, h);
  launch_gemm(stream, h, lmW, lmb, 0, out,
              8192, 512, 512, 512, 512, 512, 1, 1, 0,0,0,0,0,0, 1.f, 0, 0);
  logsoftmax512<<<dim3((unsigned)NT), dim3(128), 0, stream>>>(out);
}
